// SkipThoughts_23313082482861
// MI455X (gfx1250) — matrix operand order repaired and checked by reading, not compiled, not run on hardware
//
#include <hip/hip_runtime.h>
#include <hip/hip_bf16.h>

// ---------------------------------------------------------------------------
// GRU scan for MI455X (gfx1250): bf16 WMMA 16x16x32, fp32 accumulate.
// B=128, S=48, V=20000, E=620 (padded to 640), H=2400.
// Step kernel: B-tile staged in LDS (double-buffered, fragment-native layout,
// 48B-padded entries for conflict-free ds_load_b128), branch-free cooperative
// fill (3 x 8B chunks/thread), software-pipelined A fragments.
// ---------------------------------------------------------------------------

typedef __bf16 bf16_t;
typedef __attribute__((ext_vector_type(16))) __bf16 v16bf;
typedef __attribute__((ext_vector_type(8)))  __bf16 v8bf;
typedef __attribute__((ext_vector_type(8)))  float  v8f;

#define GB  128      // batch
#define GS  48       // seq len
#define GV  20000    // vocab
#define GE  620      // embed dim
#define GEP 640      // embed dim padded to multiple of 32
#define GH  2400     // hidden dim (divisible by 32)

// LDS B-slot geometry: entry(g, j, lane) holds 16 bf16 (32B of data) padded to
// 48B (24 elements) so lane banks = 12*lane mod 64 -> conflict-free b128 reads.
#define ENT_ELEMS 24
#define SLOT_ELEMS (3 * 2 * 32 * ENT_ELEMS)   // 4608 elems = 9216 B per slot

__device__ __forceinline__ bf16_t f2bf(float f) {
  unsigned u = __builtin_bit_cast(unsigned, f);
  u = (u + 0x7FFFu + ((u >> 16) & 1u)) >> 16;
  unsigned short s = (unsigned short)u;
  return __builtin_bit_cast(bf16_t, s);
}

// A-fragment (16x32, MxK) per CDNA5 ISA layout:
//   lanes 0-15: row M = m0+lane, K {k..k+7, k+16..k+23}
//   lanes 16-31: row M = m0+lane-16, K {k+8..k+15, k+24..k+31}
__device__ __forceinline__ v16bf load_a_frag(const bf16_t* __restrict__ base,
                                             long lda, int m0, int k0, int lane) {
  int row = m0 + (lane & 15);
  int k   = k0 + ((lane >> 4) ? 8 : 0);
  const bf16_t* p = base + (long)row * lda + k;
  v8bf lo = *(const v8bf*)(p);
  v8bf hi = *(const v8bf*)(p + 16);
  return __builtin_shufflevector(lo, hi, 0,1,2,3,4,5,6,7,8,9,10,11,12,13,14,15);
}

#define WMMA_BF16(a, b, c) \
  __builtin_amdgcn_wmma_f32_16x16x32_bf16(false, (a), false, (b), (short)0, (c), false, false)

// ---------------- LDS B-tile staging (fragment-native layout) --------------
// Fragment semantics (B 32x16, KxN): lane l of entry (g,j) holds
//   W_g[n0 + 16j + (l&15)][k + 8*(l>>4) .. +8) then [k + 16 + 8*(l>>4) .. +8)
// Tile data = 3 gates x 64 entries x 32B. Cooperative fill: 256 threads x
// 3 chunks of 8B; thread t's chunk q always belongs to gate q and all three
// share one (n,k) address -> fully branch-free, single address computation.

__device__ __forceinline__ void fetch_b_tile(const bf16_t* __restrict__ W0,
                                             const bf16_t* __restrict__ W1,
                                             const bf16_t* __restrict__ W2,
                                             long ldw, int n0, int k, int tid,
                                             uint2& d0, uint2& d1, uint2& d2) {
  int e      = tid >> 2;          // local entry 0..63 (= j*32 + lane)
  int part   = tid & 3;           // 8B chunk within the 32B entry payload
  int j      = (e >> 5) & 1;
  int lane_e = e & 31;
  int n  = n0 + 16 * j + (lane_e & 15);
  long kk = (long)k + 8 * (lane_e >> 4) + (part & 1) * 4 + (part >> 1) * 16;
  long off = (long)n * ldw + kk;
  d0 = *(const uint2*)(W0 + off);
  d1 = *(const uint2*)(W1 + off);
  d2 = *(const uint2*)(W2 + off);
}

__device__ __forceinline__ void store_b_tile(bf16_t* slot, int tid,
                                             uint2 d0, uint2 d1, uint2 d2) {
  int e    = tid >> 2;
  int part = tid & 3;
  bf16_t* p = slot + e * ENT_ELEMS + part * 4;
  *(uint2*)(p)                       = d0;   // gate 0: entries [0,64)
  *(uint2*)(p +  64 * ENT_ELEMS)     = d1;   // gate 1: entries [64,128)
  *(uint2*)(p + 128 * ENT_ELEMS)     = d2;   // gate 2: entries [128,192)
}

__device__ __forceinline__ v16bf read_b_lds(const bf16_t* slot, int g, int j, int lane) {
  const bf16_t* p = slot + (g * 64 + j * 32 + lane) * ENT_ELEMS;
  v8bf lo = *(const v8bf*)(p);
  v8bf hi = *(const v8bf*)(p + 8);
  return __builtin_shufflevector(lo, hi, 0,1,2,3,4,5,6,7,8,9,10,11,12,13,14,15);
}

// --------------------------- prep kernels ----------------------------------

__global__ void cvt_pad_kernel(const float* __restrict__ src, bf16_t* __restrict__ dst,
                               int scols, int dcols, long total) {
  long i = (long)blockIdx.x * blockDim.x + threadIdx.x;
  if (i >= total) return;
  long r = i / dcols;
  int  c = (int)(i - r * dcols);
  float v = (c < scols) ? src[r * scols + c] : 0.0f;
  dst[i] = f2bf(v);
}

__global__ void embed_gather_kernel(const int* __restrict__ tokens,
                                    const float* __restrict__ emb,
                                    bf16_t* __restrict__ X, long total) {
  long i = (long)blockIdx.x * blockDim.x + threadIdx.x;
  if (i >= total) return;
  long row = i / GEP;
  int  c   = (int)(i - row * GEP);
  int  tok = tokens[row];
  float v  = (c < GE) ? emb[(long)tok * GE + c] : 0.0f;
  X[i] = f2bf(v);
}

__global__ void init_h_kernel(float* __restrict__ hf, bf16_t* __restrict__ hb, int n) {
  int i = blockIdx.x * blockDim.x + threadIdx.x;
  if (i >= n) return;
  hf[i] = 0.0f;
  hb[i] = f2bf(0.0f);
}

// --------------------------- GRU step kernel -------------------------------
// grid = GH/32 WGs; 8 waves/WG; wave w owns M rows [16w,16w+16),
// WG owns N cols [32*blockIdx.x, +32). Fuses h@W_h*, x_t@W_i*, and gates.
__global__ void __launch_bounds__(256)
gru_step_kernel(const bf16_t* __restrict__ hb_in,   // [GB,GH] bf16 prev h
                const float*  __restrict__ hf_in,   // [GB,GH] f32 prev h
                const bf16_t* __restrict__ Xb,      // [GB*GS,GEP] bf16 embedded
                const bf16_t* __restrict__ Whr_b,   // [GH,GH]
                const bf16_t* __restrict__ Whi_b,
                const bf16_t* __restrict__ Whn_b,
                const bf16_t* __restrict__ Wir_b,   // [GH,GEP]
                const bf16_t* __restrict__ Wii_b,
                const bf16_t* __restrict__ Win_b,
                const float* __restrict__ b_ir,
                const float* __restrict__ b_ii,
                const float* __restrict__ b_in,
                const int*   __restrict__ lengths,
                float*  __restrict__ hf_out,
                bf16_t* __restrict__ hb_out,
                float*  __restrict__ out,           // [GB,GH]
                int t) {
  __shared__ bf16_t sB[2][SLOT_ELEMS];   // 2 x 9216 B

  const int tid  = threadIdx.x;
  const int lane = tid & 31;
  const int wave = tid >> 5;
  const int m0 = wave * 16;
  const int n0 = blockIdx.x * 32;

  v8f accr[2]  = {};   // x@Wir + h@Whr
  v8f acci[2]  = {};   // x@Wii + h@Whi
  v8f accnh[2] = {};   // h@Whn   (multiplied by r in epilogue)
  v8f accnx[2] = {};   // x@Win

  uint2 d0 = {}, d1 = {}, d2 = {};

  // ---- recurrent projections: K = GH, B staged in LDS (double-buffered) ----
  fetch_b_tile(Whr_b, Whi_b, Whn_b, GH, n0, 0, tid, d0, d1, d2);
  store_b_tile(sB[0], tid, d0, d1, d2);
  __syncthreads();
  v16bf a_cur = load_a_frag(hb_in, GH, m0, 0, lane);
  for (int i = 0; i < GH / 32; ++i) {
    const int k = i * 32;
    const bool more = (i + 1) < (GH / 32);
    if (more) fetch_b_tile(Whr_b, Whi_b, Whn_b, GH, n0, k + 32, tid, d0, d1, d2);
    v16bf a_nxt = a_cur;
    if (more) a_nxt = load_a_frag(hb_in, GH, m0, k + 32, lane);
    const bf16_t* slot = sB[i & 1];
    // batch all 6 fragment reads before the WMMAs -> partial dscnt waits
    v16bf br0 = read_b_lds(slot, 0, 0, lane);
    v16bf bi0 = read_b_lds(slot, 1, 0, lane);
    v16bf bn0 = read_b_lds(slot, 2, 0, lane);
    v16bf br1 = read_b_lds(slot, 0, 1, lane);
    v16bf bi1 = read_b_lds(slot, 1, 1, lane);
    v16bf bn1 = read_b_lds(slot, 2, 1, lane);
    accr[0]  = WMMA_BF16(a_cur, br0, accr[0]);
    acci[0]  = WMMA_BF16(a_cur, bi0, acci[0]);
    accnh[0] = WMMA_BF16(a_cur, bn0, accnh[0]);
    accr[1]  = WMMA_BF16(a_cur, br1, accr[1]);
    acci[1]  = WMMA_BF16(a_cur, bi1, acci[1]);
    accnh[1] = WMMA_BF16(a_cur, bn1, accnh[1]);
    if (more) store_b_tile(sB[(i + 1) & 1], tid, d0, d1, d2);
    __syncthreads();
    a_cur = a_nxt;
  }

  // ---- input projections for timestep t: K = GEP ----
  // X row for batch m is (m*GS + t) -> base shifted by t*GEP, stride GS*GEP
  const bf16_t* Xbase = Xb + (long)t * GEP;
  fetch_b_tile(Wir_b, Wii_b, Win_b, GEP, n0, 0, tid, d0, d1, d2);
  store_b_tile(sB[0], tid, d0, d1, d2);
  __syncthreads();
  a_cur = load_a_frag(Xbase, (long)GS * GEP, m0, 0, lane);
  for (int i = 0; i < GEP / 32; ++i) {
    const int k = i * 32;
    const bool more = (i + 1) < (GEP / 32);
    if (more) fetch_b_tile(Wir_b, Wii_b, Win_b, GEP, n0, k + 32, tid, d0, d1, d2);
    v16bf a_nxt = a_cur;
    if (more) a_nxt = load_a_frag(Xbase, (long)GS * GEP, m0, k + 32, lane);
    const bf16_t* slot = sB[i & 1];
    v16bf br0 = read_b_lds(slot, 0, 0, lane);
    v16bf bi0 = read_b_lds(slot, 1, 0, lane);
    v16bf bn0 = read_b_lds(slot, 2, 0, lane);
    v16bf br1 = read_b_lds(slot, 0, 1, lane);
    v16bf bi1 = read_b_lds(slot, 1, 1, lane);
    v16bf bn1 = read_b_lds(slot, 2, 1, lane);
    accr[0]  = WMMA_BF16(a_cur, br0, accr[0]);
    acci[0]  = WMMA_BF16(a_cur, bi0, acci[0]);
    accnx[0] = WMMA_BF16(a_cur, bn0, accnx[0]);
    accr[1]  = WMMA_BF16(a_cur, br1, accr[1]);
    acci[1]  = WMMA_BF16(a_cur, bi1, acci[1]);
    accnx[1] = WMMA_BF16(a_cur, bn1, accnx[1]);
    if (more) store_b_tile(sB[(i + 1) & 1], tid, d0, d1, d2);
    __syncthreads();
    a_cur = a_nxt;
  }

  // ---- gate epilogue ----
  // C/D layout: lane%16 = N, VGPR v = M=v (lanes 0-15) / M=v+8 (lanes 16-31)
  const int laneLo = lane & 15;
  const int hsel   = (lane >> 4) * 8;
#pragma unroll
  for (int j = 0; j < 2; ++j) {
    int n = n0 + 16 * j + laneLo;
    float br_ = b_ir[n];
    float bi_ = b_ii[n];
    float bn_ = b_in[n];
#pragma unroll
    for (int r = 0; r < 8; ++r) {
      int m = m0 + r + hsel;
      long idx = (long)m * GH + n;
      float hp = hf_in[idx];
      float rg = 1.0f / (1.0f + __expf(-(accr[j][r] + br_)));
      float ig = 1.0f / (1.0f + __expf(-(acci[j][r] + bi_)));
      float xa = accnx[j][r] + bn_ + rg * accnh[j][r];
      xa = fminf(fmaxf(xa, -15.0f), 15.0f);
      float e2 = __expf(2.0f * xa);
      float ng = (e2 - 1.0f) / (e2 + 1.0f);     // tanh
      float hn = (1.0f - ig) * ng + ig * hp;
      hf_out[idx] = hn;
      hb_out[idx] = f2bf(hn);
      if (lengths[m] == t + 1) out[idx] = hn;    // last valid hidden state
    }
  }
}

// --------------------------- host launcher ---------------------------------

extern "C" void kernel_launch(void* const* d_in, const int* in_sizes, int n_in,
                              void* d_out, int out_size, void* d_ws, size_t ws_size,
                              hipStream_t stream) {
  (void)in_sizes; (void)n_in; (void)out_size; (void)ws_size;

  const int*   tokens  = (const int*)  d_in[0];
  const int*   lengths = (const int*)  d_in[1];
  const float* emb     = (const float*)d_in[2];
  const float* W_ir    = (const float*)d_in[3];
  const float* W_ii    = (const float*)d_in[4];
  const float* W_in    = (const float*)d_in[5];
  const float* b_ir    = (const float*)d_in[6];
  const float* b_ii    = (const float*)d_in[7];
  const float* b_in    = (const float*)d_in[8];
  const float* W_hr    = (const float*)d_in[9];
  const float* W_hi    = (const float*)d_in[10];
  const float* W_hn    = (const float*)d_in[11];
  float* out = (float*)d_out;

  char* ws = (char*)d_ws;
  size_t off = 0;
  auto alloc = [&](size_t bytes) -> char* {
    char* p = ws + off;
    off = (off + bytes + 255) & ~(size_t)255;
    return p;
  };

  const size_t szWi = (size_t)GH * GEP * sizeof(bf16_t);
  const size_t szWh = (size_t)GH * GH  * sizeof(bf16_t);
  const size_t szX  = (size_t)GB * GS * GEP * sizeof(bf16_t);
  const size_t szHf = (size_t)GB * GH * sizeof(float);
  const size_t szHb = (size_t)GB * GH * sizeof(bf16_t);

  bf16_t* Wir_b = (bf16_t*)alloc(szWi);
  bf16_t* Wii_b = (bf16_t*)alloc(szWi);
  bf16_t* Win_b = (bf16_t*)alloc(szWi);
  bf16_t* Whr_b = (bf16_t*)alloc(szWh);
  bf16_t* Whi_b = (bf16_t*)alloc(szWh);
  bf16_t* Whn_b = (bf16_t*)alloc(szWh);
  bf16_t* Xb    = (bf16_t*)alloc(szX);
  float*  hf0   = (float*) alloc(szHf);
  float*  hf1   = (float*) alloc(szHf);
  bf16_t* hb0   = (bf16_t*)alloc(szHb);
  bf16_t* hb1   = (bf16_t*)alloc(szHb);

  const int TPB = 256;

  {
    long n = (long)GH * GEP;
    dim3 g((unsigned)((n + TPB - 1) / TPB));
    cvt_pad_kernel<<<g, TPB, 0, stream>>>(W_ir, Wir_b, GE, GEP, n);
    cvt_pad_kernel<<<g, TPB, 0, stream>>>(W_ii, Wii_b, GE, GEP, n);
    cvt_pad_kernel<<<g, TPB, 0, stream>>>(W_in, Win_b, GE, GEP, n);
  }
  {
    long n = (long)GH * GH;
    dim3 g((unsigned)((n + TPB - 1) / TPB));
    cvt_pad_kernel<<<g, TPB, 0, stream>>>(W_hr, Whr_b, GH, GH, n);
    cvt_pad_kernel<<<g, TPB, 0, stream>>>(W_hi, Whi_b, GH, GH, n);
    cvt_pad_kernel<<<g, TPB, 0, stream>>>(W_hn, Whn_b, GH, GH, n);
  }
  {
    long n = (long)GB * GS * GEP;
    dim3 g((unsigned)((n + TPB - 1) / TPB));
    embed_gather_kernel<<<g, TPB, 0, stream>>>(tokens, emb, Xb, n);
  }
  {
    int n = GB * GH;
    dim3 g((unsigned)((n + TPB - 1) / TPB));
    init_h_kernel<<<g, TPB, 0, stream>>>(hf0, hb0, n);
  }

  float*  hf[2] = {hf0, hf1};
  bf16_t* hb[2] = {hb0, hb1};
  int cur = 0;
  for (int t = 0; t < GS; ++t) {
    gru_step_kernel<<<dim3(GH / 32), 256, 0, stream>>>(
        hb[cur], hf[cur], Xb,
        Whr_b, Whi_b, Whn_b,
        Wir_b, Wii_b, Win_b,
        b_ir, b_ii, b_in, lengths,
        hf[1 - cur], hb[1 - cur], out, t);
    cur ^= 1;
  }
}
